// Heads_86998857548303
// MI455X (gfx1250) — compile-verified, matrix operand order checked by reading, not run on hardware
//
#include <hip/hip_runtime.h>
#include <hip/hip_bf16.h>

// MI455X / gfx1250, wave32. All GEMM-shaped math via v_wmma_f32_16x16x32_bf16.
// A-tiles staged into LDS by the Tensor Data Mover (tensor_load_to_lds +
// s_wait_tensorcnt), double-buffered so DMA overlaps the wmma stream.

typedef __bf16 bf16_t;
typedef __attribute__((ext_vector_type(16))) __bf16 v16bf;
typedef __attribute__((ext_vector_type(8)))  float  v8f;
typedef unsigned int u32;
typedef __attribute__((ext_vector_type(4))) u32 v4u;
typedef __attribute__((ext_vector_type(8))) int v8i;
typedef __attribute__((ext_vector_type(4))) int v4i;

#define WMMA_BF16(a, b, c) \
  __builtin_amdgcn_wmma_f32_16x16x32_bf16(false, (a), false, (b), (short)0, (c), false, false)

#if defined(__has_builtin)
#if __has_builtin(__builtin_amdgcn_tensor_load_to_lds) && \
    __has_builtin(__builtin_amdgcn_s_wait_tensorcnt)
#define USE_TDM 1
#endif
#endif
#ifndef USE_TDM
#define USE_TDM 0
#endif

union FragU { v16bf v; uint4 q[2]; };

// Load a 16(row) x 32(K) bf16 fragment in the CDNA5 WMMA 16-bit A/B register
// layout from a row-major matrix (leading dim ld, elements).
//   lane L: row = row0 + (L&15), khalf = L>>4
//   q[0] -> K = k0 + khalf*8 + [0..7]      (VGPR0..3, 2 bf16 each)
//   q[1] -> K = k0 + 16 + khalf*8 + [0..7] (VGPR4..7)
__device__ __forceinline__ v16bf load_frag16x32(const bf16_t* __restrict__ base,
                                                int ld, int row0, int k0) {
  const int lane  = threadIdx.x & 31;
  const int r     = row0 + (lane & 15);
  const int khalf = lane >> 4;
  const bf16_t* p = base + (size_t)r * (size_t)ld + (size_t)(k0 + khalf * 8);
  FragU u;
  u.q[0] = *(const uint4*)(p);
  u.q[1] = *(const uint4*)(p + 16);
  return u.v;
}

#if USE_TDM
// TDM: DMA a 32x32 bf16 tile (rows x K) of a row-major [rows_total, K] tensor
// into LDS at lds_byte_off, rows stored contiguously (32 elements each).
// D# layout per CDNA5 ISA.
__device__ __forceinline__ void tdm_stage_a32x32(const bf16_t* gsrc, u32 lds_byte_off,
                                                 int K, int rows_total) {
  const unsigned long long ga = (unsigned long long)(size_t)gsrc;
  v4u g0;
  g0.x = 1u;                                           // count=1, user D#
  g0.y = lds_byte_off;                                 // lds_addr
  g0.z = (u32)(ga & 0xffffffffu);                      // global_addr[31:0]
  g0.w = (u32)((ga >> 32) & 0x01ffffffu) | (2u << 30); // global_addr[56:32] | type=2
  const u32 td0 = (u32)K, td1 = (u32)rows_total;
  const u32 tile0 = 32u, tile1 = 32u;
  const unsigned long long s0 = (unsigned long long)(u32)K; // dim0 stride (elems)
  v8i g1;
  g1[0] = (int)(1u << 16);                                  // data_size=1 (2B)
  g1[1] = (int)((td0 & 0xffffu) << 16);                     // tensor_dim0[15:0]
  g1[2] = (int)((td0 >> 16) | ((td1 & 0xffffu) << 16));     // td0 hi | td1 lo
  g1[3] = (int)((td1 >> 16) | (tile0 << 16));               // td1 hi | tile_dim0
  g1[4] = (int)tile1;                                       // tile_dim1 (tile_dim2=0)
  g1[5] = (int)(u32)(s0 & 0xffffffffu);                     // dim0_stride[31:0]
  g1[6] = (int)(u32)((s0 >> 32) & 0xffffu);                 // dim0_stride[47:32]
  g1[7] = 0;
  const v4i gz = {0, 0, 0, 0};
#if __clang_major__ >= 23
  const v8i gz8 = {0, 0, 0, 0, 0, 0, 0, 0};
  __builtin_amdgcn_tensor_load_to_lds(g0, g1, gz, gz, gz8, 0);
#else
  __builtin_amdgcn_tensor_load_to_lds(g0, g1, gz, gz, 0);
#endif
}
#endif

// ---------------------------------------------------------------------------
// Conversion / repacking kernels (bandwidth trivial vs. the GEMMs)
// ---------------------------------------------------------------------------
__global__ void cvt_f32_bf16(const float* __restrict__ in, bf16_t* __restrict__ out, int n) {
  int i = blockIdx.x * blockDim.x + threadIdx.x;
  if (i < n) out[i] = (bf16_t)in[i];
}

// W: [H, D, HS] f32  ->  Wt: [H*HS, D] bf16 (B^T form: row = output feature n)
__global__ void cvt_headw(const float* __restrict__ W, bf16_t* __restrict__ Wt,
                          int D, int HS, int n_total) {
  int i = blockIdx.x * blockDim.x + threadIdx.x;
  if (i >= n_total) return;
  int d = i % D;
  int n = i / D;
  int h = n / HS, e = n % HS;
  Wt[i] = (bf16_t)W[((size_t)h * D + d) * HS + e];
}

// Wo: [D, D] f32 (in-feature major) -> WoT: [D(out), D(in)] bf16
__global__ void cvt_wo(const float* __restrict__ W, bf16_t* __restrict__ Wt,
                       int D, int n_total) {
  int i = blockIdx.x * blockDim.x + threadIdx.x;
  if (i >= n_total) return;
  int k = i % D;
  int n = i / D;
  Wt[i] = (bf16_t)W[(size_t)k * D + n];
}

// ---------------------------------------------------------------------------
// WMMA GEMM: C[M,N] = A[M,K] * Bt[N,K]^T.
// Block = 256 threads (8 waves) -> 32(M) x 512(N); wave -> 32x64 (8 C tiles).
// The 32-row A tile is TDM-staged into LDS (double-buffered) and shared by
// all 8 waves; B fragments stream from global (weights live in the 192MB L2)
// and are reused across both M-subtiles.
// MODE 0: store bf16 as [B,H,S,64]   (Q, K projections)
// MODE 1: store bf16 as [B,H,64,S]   (V transposed, for the PV wmma B-frags)
// MODE 2: store f32 row-major + bias (output projection -> d_out)
// ---------------------------------------------------------------------------
template <int MODE>
__global__ void __launch_bounds__(256)
gemm32x512_wmma(const bf16_t* __restrict__ A, const bf16_t* __restrict__ Bt,
                bf16_t* __restrict__ outB, float* __restrict__ outF,
                const float* __restrict__ bias, int M, int N, int K, int S, int H) {
  __shared__ __align__(16) bf16_t Asm[2][32 * 32];   // double-buffered A tile
  const int wid  = threadIdx.x >> 5;
  const int lane = threadIdx.x & 31;
  const int nblkcols = N >> 9;                       // N / 512
  const int mblk = blockIdx.x / nblkcols;
  const int nblk = blockIdx.x % nblkcols;
  const int m0 = mblk * 32;
  const int n0 = nblk * 512 + wid * 64;
  const int nk = K >> 5;

  const v8f z = {0.f, 0.f, 0.f, 0.f, 0.f, 0.f, 0.f, 0.f};
  v8f acc[2][4] = {{z, z, z, z}, {z, z, z, z}};

#if USE_TDM
  if (wid == 0)
    tdm_stage_a32x32(A + (size_t)m0 * K, (u32)(size_t)&Asm[0][0], K, M);
#endif

  for (int ks = 0; ks < nk; ++ks) {
    const int cur = ks & 1;
#if USE_TDM
    if (wid == 0) {
      if (ks + 1 < nk) {
        // overlap: kick off next tile's DMA, then wait for the current one
        tdm_stage_a32x32(A + (size_t)m0 * K + (size_t)(ks + 1) * 32,
                         (u32)(size_t)&Asm[cur ^ 1][0], K, M);
        __builtin_amdgcn_s_wait_tensorcnt(1);
      } else {
        __builtin_amdgcn_s_wait_tensorcnt(0);
      }
    }
    __syncthreads();   // publish TDM-written LDS tile to all 8 waves
#else
    {
      // fallback: cooperative global->LDS copy (8B per thread = 2KB)
      const int t = threadIdx.x, rr = t >> 3, cc = (t & 7) * 4;
      *(uint2*)&Asm[cur][rr * 32 + cc] =
          *(const uint2*)&A[(size_t)(m0 + rr) * K + (size_t)ks * 32 + cc];
    }
    __syncthreads();
#endif

    const int k0 = ks * 32;
    const v16bf af0 = load_frag16x32(&Asm[cur][0], 32, 0, 0);
    const v16bf af1 = load_frag16x32(&Asm[cur][0], 32, 16, 0);
#pragma unroll
    for (int nt = 0; nt < 4; ++nt) {
      const v16bf bf = load_frag16x32(Bt, K, n0 + nt * 16, k0);
      acc[0][nt] = WMMA_BF16(af0, bf, acc[0][nt]);
      acc[1][nt] = WMMA_BF16(af1, bf, acc[1][nt]);
    }
    __syncthreads();   // all waves done reading Asm[cur] before it is re-filled
  }

  // C/D layout: VGPR r holds M = r + 8*(lane>=16), N = lane&15 (per 16-col tile)
  const int half = lane >> 4, nl = lane & 15;
#pragma unroll
  for (int mt = 0; mt < 2; ++mt) {
#pragma unroll
    for (int nt = 0; nt < 4; ++nt) {
      const int n = n0 + nt * 16 + nl;
#pragma unroll
      for (int r = 0; r < 8; ++r) {
        const int m = m0 + mt * 16 + r + 8 * half;
        const float v = acc[mt][nt][r];
        if (MODE == 2) {
          outF[(size_t)m * N + n] = v + bias[n];
        } else {
          const int b = m / S, s = m % S, h = n >> 6, e = n & 63;
          size_t addr;
          if (MODE == 0) addr = (((size_t)(b * H + h) * S + s) << 6) + (size_t)e; // [B,H,S,64]
          else           addr = (((size_t)(b * H + h) << 6) + (size_t)e) * S + s; // [B,H,64,S]
          outB[addr] = (bf16_t)v;
        }
      }
    }
  }
}

// ---------------------------------------------------------------------------
// Flash attention: one wave per (b, h, 16-row query block).
// Per 32-key block: 4 wmma (scores, K-dim=64), online softmax in f32 using the
// C-tile layout, P -> LDS -> A-fragment repack, 4 wmma (P*V, 4 e-tiles).
// ---------------------------------------------------------------------------
__global__ void __launch_bounds__(256)
attn_fwd_wmma(const bf16_t* __restrict__ Q,   // [B,H,S,64]
              const bf16_t* __restrict__ Km,  // [B,H,S,64]
              const bf16_t* __restrict__ Vt,  // [B,H,64,S]
              bf16_t* __restrict__ O,         // [B*S, H*64] bf16
              int Bn, int H, int S, float scale) {
  (void)Bn;
  __shared__ __align__(16) bf16_t pstage[8][16 * 32];  // per-wave 16x32 P tile
  const int wslot = threadIdx.x >> 5;
  const int lane  = threadIdx.x & 31;
  const int wave  = blockIdx.x * 8 + wslot;
  const int qblocks = S >> 4;
  const int qblk = wave % qblocks;
  const int h    = (wave / qblocks) % H;
  const int b    = wave / (qblocks * H);
  const int q0   = qblk << 4;

  const bf16_t* Qb = Q  + ((size_t)(b * H + h) * S << 6);
  const bf16_t* Kb = Km + ((size_t)(b * H + h) * S << 6);
  const bf16_t* Vb = Vt + ((size_t)(b * H + h) * S << 6);  // 64 x S

  const v16bf qf0 = load_frag16x32(Qb, 64, q0, 0);
  const v16bf qf1 = load_frag16x32(Qb, 64, q0, 32);

  const v8f z = {0.f, 0.f, 0.f, 0.f, 0.f, 0.f, 0.f, 0.f};
  v8f acc[4] = {z, z, z, z};
  float rowmax[8], rowsum[8];
#pragma unroll
  for (int r = 0; r < 8; ++r) { rowmax[r] = -__builtin_inff(); rowsum[r] = 0.f; }

  const int half = lane >> 4, nl = lane & 15;
  const int nj = (q0 >> 5) + 1;  // causal: only key blocks with keys <= q0+15

  for (int jb = 0; jb < nj; ++jb) {
    const int j0 = jb << 5;

    // scores: S[16x32] = Q[16x64] @ K[j0..j0+31]^T  (two 16-col C tiles)
    v8f s0 = z, s1 = z;
    {
      v16bf kf0 = load_frag16x32(Kb, 64, j0,      0);
      v16bf kf1 = load_frag16x32(Kb, 64, j0 + 16, 0);
      s0 = WMMA_BF16(qf0, kf0, s0);
      s1 = WMMA_BF16(qf0, kf1, s1);
      kf0 = load_frag16x32(Kb, 64, j0,      32);
      kf1 = load_frag16x32(Kb, 64, j0 + 16, 32);
      s0 = WMMA_BF16(qf1, kf0, s0);
      s1 = WMMA_BF16(qf1, kf1, s1);
    }

    const int t0 = j0 + nl, t1 = t0 + 16;
#pragma unroll
    for (int r = 0; r < 8; ++r) {
      const int srow = q0 + r + 8 * half;
      float a = s0[r] * scale; if (t0 > srow) a = -__builtin_inff();
      float c = s1[r] * scale; if (t1 > srow) c = -__builtin_inff();
      // row max across the 16-lane half (xor masks < 16 stay within halves)
      float mx = fmaxf(a, c);
#pragma unroll
      for (int off = 1; off < 16; off <<= 1) mx = fmaxf(mx, __shfl_xor(mx, off, 32));
      const float nmc  = fmaxf(fmaxf(rowmax[r], mx), -1e38f);
      const float corr = __expf(rowmax[r] - nmc);  // -inf old max -> 0
      rowmax[r] = nmc;
      const float p0 = __expf(a - nmc);
      const float p1 = __expf(c - nmc);
      float ps = p0 + p1;
#pragma unroll
      for (int off = 1; off < 16; off <<= 1) ps += __shfl_xor(ps, off, 32);
      rowsum[r] = rowsum[r] * corr + ps;
#pragma unroll
      for (int nt = 0; nt < 4; ++nt) acc[nt][r] *= corr;
      // stage P (C layout -> row-major 16x32 bf16 in LDS)
      pstage[wslot][(r + 8 * half) * 32 + nl]      = (bf16_t)p0;
      pstage[wslot][(r + 8 * half) * 32 + nl + 16] = (bf16_t)p1;
    }

    // repack P as A fragment (same-wave LDS store->load is in-order)
    const v16bf pf = load_frag16x32(&pstage[wslot][0], 32, 0, 0);
#pragma unroll
    for (int nt = 0; nt < 4; ++nt) {
      v16bf vf = load_frag16x32(Vb + j0, S, nt * 16, 0);  // B^T form: rows = e
      acc[nt] = WMMA_BF16(pf, vf, acc[nt]);
    }
  }

  // normalize and store concat-head bf16 output
#pragma unroll
  for (int r = 0; r < 8; ++r) {
    const float inv = 1.0f / rowsum[r];
    const int m = q0 + r + 8 * half;
    const size_t rowbase = (size_t)(b * S + m) * (size_t)(H * 64) + (size_t)(h * 64);
#pragma unroll
    for (int nt = 0; nt < 4; ++nt)
      O[rowbase + nt * 16 + nl] = (bf16_t)(acc[nt][r] * inv);
  }
}

// ---------------------------------------------------------------------------
extern "C" void kernel_launch(void* const* d_in, const int* in_sizes, int n_in,
                              void* d_out, int out_size, void* d_ws, size_t ws_size,
                              hipStream_t stream) {
  (void)in_sizes; (void)n_in; (void)out_size; (void)ws_size;
  const int B = 4, S = 1024, D = 1024, H = 16, HS = 64;
  const int M = B * S;                       // 4096
  const float scale = 1.0f / 32.0f;          // 1/sqrt(D)

  const float* data = (const float*)d_in[0];
  const float* Wq   = (const float*)d_in[1];
  const float* Wk   = (const float*)d_in[2];
  const float* Wv   = (const float*)d_in[3];
  const float* Wo   = (const float*)d_in[4];
  const float* bo   = (const float*)d_in[5];
  float* out        = (float*)d_out;

  // workspace layout (bf16 elements): 24M elems = 48 MB
  bf16_t* ws  = (bf16_t*)d_ws;
  bf16_t* Xbf = ws;                               // [M, D]
  bf16_t* WqT = Xbf + (size_t)M * D;              // [D, D] (B^T form)
  bf16_t* WkT = WqT + (size_t)D * D;
  bf16_t* WvT = WkT + (size_t)D * D;
  bf16_t* WoT = WvT + (size_t)D * D;
  bf16_t* Qm  = WoT + (size_t)D * D;              // [B,H,S,64]
  bf16_t* Km  = Qm  + (size_t)M * D;
  bf16_t* Vt  = Km  + (size_t)M * D;              // [B,H,64,S]
  bf16_t* AO  = Vt  + (size_t)M * D;              // [M, D] concat heads

  // converts
  cvt_f32_bf16<<<(M * D) / 256, 256, 0, stream>>>(data, Xbf, M * D);
  cvt_headw<<<(D * D) / 256, 256, 0, stream>>>(Wq, WqT, D, HS, D * D);
  cvt_headw<<<(D * D) / 256, 256, 0, stream>>>(Wk, WkT, D, HS, D * D);
  cvt_headw<<<(D * D) / 256, 256, 0, stream>>>(Wv, WvT, D, HS, D * D);
  cvt_wo<<<(D * D) / 256, 256, 0, stream>>>(Wo, WoT, D, D * D);

  // QKV projections: blocks of 8 waves; block tile 32x512
  const int gemmBlocks = (M / 32) * (D / 512);    // 256
  gemm32x512_wmma<0><<<gemmBlocks, 256, 0, stream>>>(Xbf, WqT, Qm, nullptr, nullptr, M, D, D, S, H);
  gemm32x512_wmma<0><<<gemmBlocks, 256, 0, stream>>>(Xbf, WkT, Km, nullptr, nullptr, M, D, D, S, H);
  gemm32x512_wmma<1><<<gemmBlocks, 256, 0, stream>>>(Xbf, WvT, Vt, nullptr, nullptr, M, D, D, S, H);

  // causal flash attention
  const int attnWaves = B * H * (S / 16);         // 4096
  attn_fwd_wmma<<<attnWaves / 8, 256, 0, stream>>>(Qm, Km, Vt, AO, B, H, S, scale);

  // output projection + bias -> f32
  gemm32x512_wmma<2><<<gemmBlocks, 256, 0, stream>>>(AO, WoT, nullptr, out, bo, M, D, D, S, H);
}
